// GNNRecommender_57019985822064
// MI455X (gfx1250) — matrix operand order repaired and checked by reading, not compiled, not run on hardware
//
#include <hip/hip_runtime.h>

typedef _Float16 h16;
typedef __attribute__((ext_vector_type(16))) _Float16 v16h;
typedef __attribute__((ext_vector_type(8)))  _Float16 v8h;
typedef __attribute__((ext_vector_type(8)))  float    v8f;

#define N_NODES 50000
#define N_EDGES 800000
#define IN_DIM  128
#define HID_DIM 128
#define OUT_DIM 64

// ---------------------------------------------------------------------------
// Degree / norm preparation
// ---------------------------------------------------------------------------
__global__ void set_deg_one(float* __restrict__ deg, int n) {
    int i = blockIdx.x * blockDim.x + threadIdx.x;
    if (i < n) deg[i] = 1.0f;                       // self-loop weight
}

__global__ void accum_deg(const int* __restrict__ dst, const float* __restrict__ ew,
                          float* __restrict__ deg, int e) {
    int i = blockIdx.x * blockDim.x + threadIdx.x;
    if (i < e)
        __hip_atomic_fetch_add(&deg[dst[i]], ew[i], __ATOMIC_RELAXED,
                               __HIP_MEMORY_SCOPE_AGENT);
}

__global__ void compute_dinv(const float* __restrict__ deg, float* __restrict__ dinv, int n) {
    int i = blockIdx.x * blockDim.x + threadIdx.x;
    if (i < n) {
        float d = deg[i];
        dinv[i] = d > 0.0f ? __frsqrt_rn(d) : 0.0f;
    }
}

// ---------------------------------------------------------------------------
// Elementwise converts
// ---------------------------------------------------------------------------
__global__ void cvt_f32_to_f16(const float* __restrict__ in, h16* __restrict__ out, int n) {
    int i = blockIdx.x * blockDim.x + threadIdx.x;
    if (i < n) out[i] = (h16)in[i];
}

__global__ void relu_cvt_f16(const float* __restrict__ in, h16* __restrict__ out, int n) {
    int i = blockIdx.x * blockDim.x + threadIdx.x;
    if (i < n) {
        float v = in[i];
        out[i] = (h16)(v > 0.0f ? v : 0.0f);
    }
}

// Repack weight [K,N] f32 row-major into WMMA B-fragment order:
// linear p = ((kb*(N/16)+nb)*32 + lane)*16 + e,
// value  = W[kb*32 + (lane>>4)*16 + e, nb*16 + (lane&15)]
__global__ void repack_w_f16(const float* __restrict__ W, h16* __restrict__ Bp,
                             int K, int N) {
    int p = blockIdx.x * blockDim.x + threadIdx.x;
    if (p >= K * N) return;
    int e  = p & 15;
    int L  = (p >> 4) & 31;
    int nT = N >> 4;
    int nb = (p >> 9) % nT;
    int kb = (p >> 9) / nT;
    int k  = (kb << 5) + ((L >> 4) << 4) + e;
    int n  = (nb << 4) + (L & 15);
    Bp[p] = (h16)W[(size_t)k * N + n];
}

// ---------------------------------------------------------------------------
// GEMM: C[M,N] = A[M,K](f16 row-major) x Bp(prepacked f16), f32 accumulate.
// One wave per 16x16 tile; K stepped 32 per v_wmma_f32_16x16x32_f16.
// ---------------------------------------------------------------------------
__global__ __launch_bounds__(256, 1)
void gcn_gemm_wmma(const h16* __restrict__ A, const h16* __restrict__ Bp,
                   float* __restrict__ C, int M, int N, int K) {
    const int lane  = threadIdx.x & 31;
    const int wave  = threadIdx.x >> 5;
    const int nT    = N >> 4;
    const int tiles = (M >> 4) * nT;
    const int gw    = blockIdx.x * 8 + wave;
    if (gw >= tiles) return;                       // wave-uniform: EXEC stays all-1

    const int tm  = gw / nT;
    const int tn  = gw % nT;
    const int hs  = lane >> 4;                     // half-select (K sub-block)
    const int mrw = (tm << 4) + (lane & 15);       // A row handled by this lane

    const h16* arow   = A + (size_t)mrw * K;
    const h16* bfrag  = Bp + ((size_t)tn * 32 + lane) * 16;
    const size_t bkst = (size_t)nT * 512;          // halves per K-block of Bp

    v8f acc = {};
    const int kBlocks = K >> 5;
    for (int kb = 0; kb < kBlocks; ++kb) {
        const h16* ak = arow + (kb << 5);
        // A layout (matches Bp): elems 0..15 -> K = hs*16+0..15
        v16h a = *(const v16h*)(ak + hs * 16);
        v16h b = *(const v16h*)(bfrag + kb * bkst);
        acc = __builtin_amdgcn_wmma_f32_16x16x32_f16(
                  /*neg_a=*/false, a, /*neg_b=*/false, b,
                  /*c_mod=*/(short)0, acc, /*reuse_a=*/false, /*reuse_b=*/false);
    }
    // C/D layout: VGPR r -> row (hs*8 + r), col = lane&15
    float* crow = C + (size_t)((tm << 4) + (hs << 3)) * N + (tn << 4) + (lane & 15);
#pragma unroll
    for (int r = 0; r < 8; ++r) crow[(size_t)r * N] = acc[r];
}

// ---------------------------------------------------------------------------
// Aggregation: init = self-loop + bias, then per-edge atomic scatter
// ---------------------------------------------------------------------------
__global__ void init_agg(const float* __restrict__ h, const float* __restrict__ dinv,
                         const float* __restrict__ bias, float* __restrict__ agg,
                         int nNodes, int logF) {
    int i = blockIdx.x * blockDim.x + threadIdx.x;
    int node = i >> logF;
    if (node >= nNodes) return;
    int f = i & ((1 << logF) - 1);
    float di = dinv[node];
    agg[i] = h[i] * di * di + bias[f];
}

__global__ void edge_scatter(const int* __restrict__ src, const int* __restrict__ dst,
                             const float* __restrict__ ew, const float* __restrict__ dinv,
                             const float* __restrict__ h, float* __restrict__ agg,
                             int nEdges, int logF) {
    long long tid = (long long)blockIdx.x * blockDim.x + threadIdx.x;
    int e = (int)(tid >> logF);
    if (e >= nEdges) return;
    int f = (int)tid & ((1 << logF) - 1);
    int s = src[e], d = dst[e];
    float norm = dinv[s] * ew[e] * dinv[d];
    float v = h[((size_t)s << logF) + f] * norm;
    __hip_atomic_fetch_add(&agg[((size_t)d << logF) + f], v, __ATOMIC_RELAXED,
                           __HIP_MEMORY_SCOPE_AGENT);
}

// ---------------------------------------------------------------------------
// Host-side orchestration
// ---------------------------------------------------------------------------
static inline size_t alignup(size_t x) { return (x + 255) & ~(size_t)255; }
static inline int cdiv(long long a, long long b) { return (int)((a + b - 1) / b); }

extern "C" void kernel_launch(void* const* d_in, const int* in_sizes, int n_in,
                              void* d_out, int out_size, void* d_ws, size_t ws_size,
                              hipStream_t stream) {
    const float* x  = (const float*)d_in[0];
    const int*   ei = (const int*)d_in[1];
    const float* ew = (const float*)d_in[2];
    const float* W1 = (const float*)d_in[3];
    const float* b1 = (const float*)d_in[4];
    const float* W2 = (const float*)d_in[5];
    const float* b2 = (const float*)d_in[6];
    float* out = (float*)d_out;

    const int N = N_NODES, E = N_EDGES;
    const int* src = ei;
    const int* dst = ei + E;

    // workspace carve
    char* ws = (char*)d_ws;
    float* deg  = (float*)ws;  ws += alignup((size_t)N * 4);
    float* dinv = (float*)ws;  ws += alignup((size_t)N * 4);
    h16*   xh   = (h16*)ws;    ws += alignup((size_t)N * IN_DIM * 2);   // reused as z_f16
    h16*   W1p  = (h16*)ws;    ws += alignup((size_t)IN_DIM * HID_DIM * 2);
    h16*   W2p  = (h16*)ws;    ws += alignup((size_t)HID_DIM * OUT_DIM * 2);
    float* h1   = (float*)ws;  ws += alignup((size_t)N * HID_DIM * 4);  // reused as h2
    float* agg1 = (float*)ws;  ws += alignup((size_t)N * HID_DIM * 4);
    h16*   zh   = xh;
    float* h2   = h1;

    const int B = 256;

    // normalization terms (shared by both layers)
    set_deg_one<<<cdiv(N, B), B, 0, stream>>>(deg, N);
    accum_deg<<<cdiv(E, B), B, 0, stream>>>(dst, ew, deg, E);
    compute_dinv<<<cdiv(N, B), B, 0, stream>>>(deg, dinv, N);

    // f16 staging
    cvt_f32_to_f16<<<cdiv((long long)N * IN_DIM, B), B, 0, stream>>>(x, xh, N * IN_DIM);
    repack_w_f16<<<cdiv(IN_DIM * HID_DIM, B), B, 0, stream>>>(W1, W1p, IN_DIM, HID_DIM);
    repack_w_f16<<<cdiv(HID_DIM * OUT_DIM, B), B, 0, stream>>>(W2, W2p, HID_DIM, OUT_DIM);

    // ---- layer 1 ----
    {
        int tiles = (N >> 4) * (HID_DIM >> 4);
        gcn_gemm_wmma<<<cdiv(tiles, 8), B, 0, stream>>>(xh, W1p, h1, N, HID_DIM, IN_DIM);
    }
    init_agg<<<cdiv((long long)N * HID_DIM, B), B, 0, stream>>>(h1, dinv, b1, agg1, N, 7);
    edge_scatter<<<cdiv((long long)E * HID_DIM, B), B, 0, stream>>>(src, dst, ew, dinv,
                                                                    h1, agg1, E, 7);
    relu_cvt_f16<<<cdiv((long long)N * HID_DIM, B), B, 0, stream>>>(agg1, zh, N * HID_DIM);

    // ---- layer 2 ----
    {
        int tiles = (N >> 4) * (OUT_DIM >> 4);
        gcn_gemm_wmma<<<cdiv(tiles, 8), B, 0, stream>>>(zh, W2p, h2, N, OUT_DIM, HID_DIM);
    }
    init_agg<<<cdiv((long long)N * OUT_DIM, B), B, 0, stream>>>(h2, dinv, b2, out, N, 6);
    edge_scatter<<<cdiv((long long)E * OUT_DIM, B), B, 0, stream>>>(src, dst, ew, dinv,
                                                                    h2, out, E, 6);
    (void)in_sizes; (void)n_in; (void)out_size; (void)ws_size;
}
